// Attention_2302102471003
// MI455X (gfx1250) — compile-verified, matrix operand order checked by reading, not run on hardware
//
#include <hip/hip_runtime.h>

// Attention (LAS-style) fused kernels for MI455X / gfx1250, wave32 + WMMA bf16.
//
// Inputs (setup_inputs order):
//  0 listener_output [64,2000,512] f32
//  1 decoder_state   [64,512]      f32
//  2 lengths         [64]          i32
//  3 Wq [256,512] f32   4 bq [256] f32
//  5 Wk [256,512] f32   6 bk [256] f32
//  7 Wv [512,512] f32   8 bv [512] f32
// Outputs: context [64,512] f32, masked_attention [64,2000] f32 (concat).
//
// Workspace layout (d_ws), ~832 KB total:
//   [0        , 64KB )  q        : query [64,256] f32
//   [64KB     , 320KB)  Wk_sw    : Wk bf16, WMMA-fragment-swizzled
//   [320KB    , 832KB)  Wv_sw    : Wv bf16, WMMA-fragment-swizzled

#define BATCH 64
#define TLEN  2000
#define LDIM  512
#define DK    256
#define DC    512
#define NEG_SLOPE 0.2f

typedef __attribute__((ext_vector_type(16))) __bf16 v16bf;
typedef __attribute__((ext_vector_type(8)))  __bf16 v8bf;
typedef __attribute__((ext_vector_type(8)))  float  v8f;
typedef __attribute__((ext_vector_type(4)))  int    v4i;

#define SXS_PAD 24   // halves per (step,lane) row: 48B stride, 16B aligned, conflict-free half-wave

#if defined(__has_builtin)
#if __has_builtin(__builtin_amdgcn_global_load_async_to_lds_b128) && \
    __has_builtin(__builtin_amdgcn_s_wait_asynccnt)
#define USE_ASYNC_LDS 1
#endif
#endif

__device__ __forceinline__ float lrelu(float x) {
    return x >= 0.0f ? x : NEG_SLOPE * x;
}

// Load a 32-byte (16 x bf16) fragment as two 16B vector loads.
__device__ __forceinline__ v16bf load_frag16(const __bf16* p) {
    const v8bf* p8 = (const v8bf*)p;
    v8bf lo = p8[0];
    v8bf hi = p8[1];
    return __builtin_shufflevector(lo, hi, 0, 1, 2, 3, 4, 5, 6, 7,
                                   8, 9, 10, 11, 12, 13, 14, 15);
}

// Fragment K-index map (ISA 16-bit A 16x32 layout), h = packed element 0..15.
__device__ __forceinline__ int kmap(int h, int hiHalf) {
    return h + ((h >= 8) ? 8 : 0) + hiHalf;
}

// -------------------------------------------------------------------------
// Zero the context accumulation region.
__global__ void k_zero(float* __restrict__ p, int n) {
    int i = blockIdx.x * 256 + threadIdx.x;
    if (i < n) p[i] = 0.0f;
}

// -------------------------------------------------------------------------
// Pre-swizzle weights: W [ntiles*16, 512] f32 row-major ->
// out[((tile*16 + step)*32 + lane)*16 + h] = bf16(W[tile*16 + (lane&15)][step*32 + K(h,lane)])
// One thread per (tile, step, lane); each writes 16 contiguous halves (32B).
__global__ void __launch_bounds__(256) k_wswz(const float* __restrict__ W,
                                              __bf16* __restrict__ out,
                                              int ntiles) {
    int idx = blockIdx.x * 256 + threadIdx.x;
    if (idx >= ntiles * 16 * 32) return;
    const int lane   = idx & 31;
    const int step   = (idx >> 5) & 15;
    const int tile   = idx >> 9;
    const int hiHalf = (lane >= 16) ? 8 : 0;
    const float* wrow = W + (long)(tile * 16 + (lane & 15)) * LDIM + step * 32;
    __bf16* o = out + (long)idx * 16;
#pragma unroll
    for (int h = 0; h < 16; ++h) o[h] = (__bf16)wrow[kmap(h, hiHalf)];
}

// -------------------------------------------------------------------------
// query[b,k] = leaky_relu(decoder_state[b,:] . Wq[k,:] + bq[k])
__global__ void __launch_bounds__(256) k_query(const float* __restrict__ ds,
                                               const float* __restrict__ Wq,
                                               const float* __restrict__ bq,
                                               float* __restrict__ q) {
    __shared__ float sD[LDIM];
    const int b = blockIdx.x, tid = threadIdx.x;
    for (int i = tid; i < LDIM; i += 256) sD[i] = ds[b * LDIM + i];
    __syncthreads();
    const int k = tid;
    const float* w = Wq + (long)k * LDIM;
    float s = bq[k];
#pragma unroll 8
    for (int l = 0; l < LDIM; ++l) s = fmaf(sD[l], w[l], s);
    q[b * DK + k] = lrelu(s);
}

// -------------------------------------------------------------------------
// Scatter-store one fp32 X element into the bf16 fragment-swizzled LDS tile.
__device__ __forceinline__ void swz_store(__bf16 (*sXs)[32][SXS_PAD],
                                          int r, int c, float v) {
    const int step = c >> 5;
    const int K = c & 31;
    const int hi = (K >> 3) & 1;
    const int h = (K & 7) + ((K >= 16) ? 8 : 0);
    sXs[step][r + (hi << 4)][h] = (__bf16)v;
}

// -------------------------------------------------------------------------
// Fused key projection + energy:
// energy[b,t] = sum_k q[b,k] * leaky_relu( X[b,t,:] . Wk[k,:] + bk[k] )
// grid = (TLEN/16, BATCH), block = 256 (8 waves x 2 k-tiles = 256 k columns)
__global__ void __launch_bounds__(256) k_energy(const float* __restrict__ X,
                                                const __bf16* __restrict__ WkSw,
                                                const float* __restrict__ bk,
                                                const float* __restrict__ q,
                                                float* __restrict__ energy) {
    __shared__ __bf16 sXs[16][32][SXS_PAD];   // 24 KB, fragment-swizzled X tile
    __shared__ float  sE[16];

    const int t0  = blockIdx.x * 16;
    const int b   = blockIdx.y;
    const int tid = threadIdx.x;

    // Stage 16x512 fp32 -> bf16 swizzled LDS.
    const float* Xb = X + ((long)b * TLEN + t0) * LDIM;
    for (int i = tid; i < 16 * LDIM; i += 256) {
        int r = i >> 9, c = i & (LDIM - 1);
        swz_store(sXs, r, c, Xb[r * LDIM + c]);
    }
    if (tid < 16) sE[tid] = 0.0f;
    __syncthreads();

    const int lane   = tid & 31;
    const int wave   = tid >> 5;
    const int hiHalf = (lane >= 16) ? 8 : 0;
    const int row    = lane & 15;
    const int tile0  = wave * 2;

    v8f acc[2];
    acc[0] = (v8f){0.f, 0.f, 0.f, 0.f, 0.f, 0.f, 0.f, 0.f};
    acc[1] = (v8f){0.f, 0.f, 0.f, 0.f, 0.f, 0.f, 0.f, 0.f};

#pragma unroll 4
    for (int step = 0; step < 16; ++step) {
        v16bf a  = load_frag16(&sXs[step][lane][0]);
        v16bf b0 = load_frag16(WkSw + ((long)((tile0 + 0) * 16 + step) * 32 + lane) * 16);
        v16bf b1 = load_frag16(WkSw + ((long)((tile0 + 1) * 16 + step) * 32 + lane) * 16);
        acc[0] = __builtin_amdgcn_wmma_f32_16x16x32_bf16(
            false, a, false, b0, (short)0, acc[0], false, false);
        acc[1] = __builtin_amdgcn_wmma_f32_16x16x32_bf16(
            false, a, false, b1, (short)0, acc[1], false, false);
    }

#pragma unroll
    for (int s = 0; s < 2; ++s) {
        const int col = (tile0 + s) * 16 + row;       // lane's N column
        const float qv  = q[b * DK + col];
        const float bkv = bk[col];
        float v[8];
#pragma unroll
        for (int e = 0; e < 8; ++e) v[e] = lrelu(acc[s][e] + bkv) * qv;
        // Sum over the 16 N columns held by this half-wave.
#pragma unroll
        for (int e = 0; e < 8; ++e) {
            float x = v[e];
            x += __shfl_xor(x, 1, 16);
            x += __shfl_xor(x, 2, 16);
            x += __shfl_xor(x, 4, 16);
            x += __shfl_xor(x, 8, 16);
            v[e] = x;
        }
        if (row == 0) {
#pragma unroll
            for (int e = 0; e < 8; ++e) atomicAdd(&sE[e + hiHalf], v[e]);
        }
    }
    __syncthreads();
    if (tid < 16) energy[(long)b * TLEN + t0 + tid] = sE[tid];
}

// -------------------------------------------------------------------------
// In-place masked softmax + L1 renormalization (== reference softmax->mask->L1).
__global__ void __launch_bounds__(256) k_softmax(const int* __restrict__ lengths,
                                                 float* __restrict__ att) {
    __shared__ float sR[256];
    const int b = blockIdx.x, tid = threadIdx.x;
    float* row = att + (long)b * TLEN;
    const int len = lengths[b];

    float mx = -3.4e38f;
    for (int t = tid; t < TLEN; t += 256) mx = fmaxf(mx, row[t]);
    sR[tid] = mx;
    __syncthreads();
    for (int s = 128; s > 0; s >>= 1) {
        if (tid < s) sR[tid] = fmaxf(sR[tid], sR[tid + s]);
        __syncthreads();
    }
    mx = sR[0];
    __syncthreads();

    float sum = 0.0f;
    for (int t = tid; t < TLEN; t += 256) {
        float e = (t < len) ? __expf(row[t] - mx) : 0.0f;
        row[t] = e;
        sum += e;
    }
    sR[tid] = sum;
    __syncthreads();
    for (int s = 128; s > 0; s >>= 1) {
        if (tid < s) sR[tid] += sR[tid + s];
        __syncthreads();
    }
    const float inv = 1.0f / fmaxf(sR[0], 1e-12f);
    for (int t = tid; t < TLEN; t += 256) row[t] *= inv;
}

// -------------------------------------------------------------------------
// Raw fp32 X tile fetch into LDS: async global->LDS copy if available
// (overlaps next tile's HBM fetch with this tile's WMMA), else plain copy.
// Builtin signature (from compiler diagnostic): param 0 is v4i* (128-bit payload).
__device__ __forceinline__ void tile_fetch(const float* __restrict__ gsrc,
                                           float* __restrict__ lds_dst, int tid) {
#if defined(USE_ASYNC_LDS)
#pragma unroll
    for (int it = 0; it < 8; ++it) {
        const int off = (it * 256 + tid) * 4;   // 4 floats = 16B per lane
        __builtin_amdgcn_global_load_async_to_lds_b128(
            (v4i*)(gsrc + off), (v4i*)(lds_dst + off), 0, 0);
    }
#else
    for (int i = tid; i < 16 * LDIM; i += 256) lds_dst[i] = gsrc[i];
#endif
}

__device__ __forceinline__ void tile_wait() {
#if defined(USE_ASYNC_LDS)
    __builtin_amdgcn_s_wait_asynccnt(0);
#endif
}

// -------------------------------------------------------------------------
// Fused value projection + context:
// context[b,c] = sum_t att[b,t] * leaky_relu( X[b,t,:] . Wv[c,:] + bv[c] )
// grid = (TLEN/80, BATCH), block = 256; 5 t-tiles per block, software-pipelined:
// tile tt+1 streams into LDS (async) while tile tt runs through WMMA.
__global__ void __launch_bounds__(256) k_context(const float* __restrict__ X,
                                                 const __bf16* __restrict__ WvSw,
                                                 const float* __restrict__ bv,
                                                 const float* __restrict__ att,
                                                 float* __restrict__ ctx) {
    __shared__ float  sRaw[16 * LDIM];        // 32 KB raw fp32 tile (async target)
    __shared__ __bf16 sXs[16][32][SXS_PAD];   // 24 KB swizzled bf16 tile
    __shared__ float  sA[16];

    const int chunk = blockIdx.x;
    const int b     = blockIdx.y;
    const int tid   = threadIdx.x;
    const int lane   = tid & 31;
    const int wave   = tid >> 5;
    const int hiHalf = (lane >= 16) ? 8 : 0;
    const int row    = lane & 15;
    const int tile0  = wave * 4;

    float cacc[4] = {0.f, 0.f, 0.f, 0.f};

    const int tbase = chunk * 5 * 16;
    tile_fetch(X + ((long)b * TLEN + tbase) * LDIM, sRaw, tid);   // prologue

    for (int tt = 0; tt < 5; ++tt) {
        const int t0 = tbase + tt * 16;
        tile_wait();
        __syncthreads();   // raw tile visible to all waves; sXs free to overwrite

        // Convert raw fp32 -> swizzled bf16 fragments.
        for (int i = tid; i < 16 * LDIM; i += 256) {
            int r = i >> 9, c = i & (LDIM - 1);
            swz_store(sXs, r, c, sRaw[i]);
        }
        if (tid < 16) sA[tid] = att[(long)b * TLEN + t0 + tid];
        __syncthreads();   // sXs/sA ready; sRaw free

        if (tt < 4)        // overlap next tile's fetch with this tile's compute
            tile_fetch(X + ((long)b * TLEN + t0 + 16) * LDIM, sRaw, tid);

        v8f acc[4];
#pragma unroll
        for (int s = 0; s < 4; ++s)
            acc[s] = (v8f){0.f, 0.f, 0.f, 0.f, 0.f, 0.f, 0.f, 0.f};

#pragma unroll 4
        for (int step = 0; step < 16; ++step) {
            v16bf a = load_frag16(&sXs[step][lane][0]);
#pragma unroll
            for (int s = 0; s < 4; ++s) {
                v16bf bm = load_frag16(
                    WvSw + ((long)((tile0 + s) * 16 + step) * 32 + lane) * 16);
                acc[s] = __builtin_amdgcn_wmma_f32_16x16x32_bf16(
                    false, a, false, bm, (short)0, acc[s], false, false);
            }
        }

#pragma unroll
        for (int s = 0; s < 4; ++s) {
            const float bvv = bv[(tile0 + s) * 16 + row];
#pragma unroll
            for (int e = 0; e < 8; ++e)
                cacc[s] = fmaf(lrelu(acc[s][e] + bvv), sA[e + hiHalf], cacc[s]);
        }
    }

    // Combine M-halves (lane L and L+16 share column N = lane&15), then emit.
#pragma unroll
    for (int s = 0; s < 4; ++s) {
        float tot = cacc[s] + __shfl_xor(cacc[s], 16, 32);
        if (lane < 16)
            atomicAdd(&ctx[(long)b * DC + (tile0 + s) * 16 + lane], tot);
    }
}

// -------------------------------------------------------------------------
extern "C" void kernel_launch(void* const* d_in, const int* in_sizes, int n_in,
                              void* d_out, int out_size, void* d_ws, size_t ws_size,
                              hipStream_t stream) {
    (void)in_sizes; (void)n_in; (void)out_size; (void)ws_size;
    const float* X       = (const float*)d_in[0];
    const float* ds      = (const float*)d_in[1];
    const int*   lengths = (const int*)  d_in[2];
    const float* Wq      = (const float*)d_in[3];
    const float* bq      = (const float*)d_in[4];
    const float* Wk      = (const float*)d_in[5];
    const float* bk      = (const float*)d_in[6];
    const float* Wv      = (const float*)d_in[7];
    const float* bv      = (const float*)d_in[8];

    float* out = (float*)d_out;
    float* ctx = out;                 // [BATCH, DC]
    float* att = out + BATCH * DC;    // [BATCH, TLEN]  (energy -> attention in place)

    char* ws = (char*)d_ws;
    float*  q    = (float*) (ws);                          // 64 KB
    __bf16* WkSw = (__bf16*)(ws + 64 * 1024);              // 256 KB
    __bf16* WvSw = (__bf16*)(ws + 64 * 1024 + 256 * 1024); // 512 KB

    k_zero<<<dim3((BATCH * DC + 255) / 256), dim3(256), 0, stream>>>(ctx, BATCH * DC);
    k_wswz<<<dim3((16 * 16 * 32 + 255) / 256), dim3(256), 0, stream>>>(Wk, WkSw, 16);
    k_wswz<<<dim3((32 * 16 * 32 + 255) / 256), dim3(256), 0, stream>>>(Wv, WvSw, 32);
    k_query<<<dim3(BATCH), dim3(256), 0, stream>>>(ds, Wq, bq, q);
    k_energy<<<dim3(TLEN / 16, BATCH), dim3(256), 0, stream>>>(X, WkSw, bk, q, att);
    k_softmax<<<dim3(BATCH), dim3(256), 0, stream>>>(lengths, att);
    k_context<<<dim3(TLEN / 80, BATCH), dim3(256), 0, stream>>>(X, WvSw, bv, att, ctx);
}
